// SA_50002009260574
// MI455X (gfx1250) — compile-verified, matrix operand order checked by reading, not run on hardware
//
#include <hip/hip_runtime.h>
#include <math.h>

// Problem constants (match reference)
#define B_ 8
#define S_ 2048
#define F_ 1024
#define NEGV -1000000000.0f

typedef __attribute__((ext_vector_type(16))) __bf16 v16bf;
typedef __attribute__((ext_vector_type(8)))  float  v8f;

__device__ __forceinline__ v8f wmma_bf16(v16bf a, v16bf b, v8f c) {
    return __builtin_amdgcn_wmma_f32_16x16x32_bf16(false, a, false, b,
                                                   (short)0, c, false, false);
}

// ----- CDNA5 async global->LDS copy (ASYNCcnt), with safe fallback ---------
#if __has_builtin(__builtin_amdgcn_global_load_async_to_lds_b128)
#define USE_ASYNC 1
// Builtin expects: (v4i AS1*, v4i AS3*, imm offset, imm cpol) where v4i is a
// GCC-style vector (per the round-2 diagnostic).
typedef int v4i_g __attribute__((vector_size(16)));
typedef __attribute__((address_space(1))) v4i_g gv4;
typedef __attribute__((address_space(3))) v4i_g lv4;
#else
#define USE_ASYNC 0
#endif

__device__ __forceinline__ void async_cp16(const void* g, void* l) {
#if USE_ASYNC
    __builtin_amdgcn_global_load_async_to_lds_b128((gv4*)g, (lv4*)l, 0, 0);
#else
    *(float4*)l = *(const float4*)g;
#endif
}

__device__ __forceinline__ void wait_async() {
#if USE_ASYNC
#if __has_builtin(__builtin_amdgcn_s_wait_asynccnt)
    __builtin_amdgcn_s_wait_asynccnt(0);
#else
    asm volatile("s_wait_asynccnt 0" ::: "memory");
#endif
#endif
}

// ---------------------------------------------------------------------------
// fp32 -> bf16 conversion (grid-stride)
// ---------------------------------------------------------------------------
__global__ void sa_cvt_bf16(const float* __restrict__ in, __bf16* __restrict__ out, int n) {
    int i = blockIdx.x * blockDim.x + threadIdx.x;
    int stride = gridDim.x * blockDim.x;
    for (; i < n; i += stride) out[i] = (__bf16)in[i];
}

// ---------------------------------------------------------------------------
// Linear projection: out[n,o] = sum_f A[n,f]*W[o,f] + bias[o]
// Workgroup tile 64x128, double-buffered async global->LDS staging (BK=32).
// 8 waves: wave (wm, wn) computes a 16x64 strip = 4 WMMA accumulators,
// A fragment reused 4x from LDS.
// transpose_out: store as [b, o, s] (for V) instead of [n, o].
// ---------------------------------------------------------------------------
#define BM 64
#define BN 128
#define BK 32

__global__ __launch_bounds__(256) void sa_wmma_linear(
        const __bf16* __restrict__ A, const __bf16* __restrict__ W,
        const float* __restrict__ bias, __bf16* __restrict__ out,
        int K, int N, int transpose_out, int seq) {
    __shared__ alignas(64) __bf16 Ab[2][BM * BK];   // 4 KB per buffer
    __shared__ alignas(64) __bf16 Bb[2][BN * BK];   // 8 KB per buffer

    const int tid  = threadIdx.x;
    const int lane = tid & 31;
    const int wave = tid >> 5;     // 0..7
    const int wm   = wave & 3;     // M sub-tile (16 rows each)
    const int wn   = wave >> 2;    // N half (64 cols each)
    const int m0   = blockIdx.y * BM;
    const int n0   = blockIdx.x * BN;
    const int klane = (lane >> 4) << 4;

    // staging assignments: A tile 64x32 bf16 (4KB) -> 256 thr x 16B
    const int ar = tid >> 2;             // 0..63
    const int ac = (tid & 3) * 8;        // bf16 col: 0,8,16,24
    // B tile 128x32 bf16 (8KB) -> 256 thr x 2x16B
    const int br = tid >> 1;             // 0..127
    const int bc = (tid & 1) * 16;       // bf16 col: 0 or 16

    const __bf16* agp = A + (size_t)(m0 + ar) * K + ac;
    const __bf16* wgp = W + (size_t)(n0 + br) * K + bc;

    auto stage = [&](int buf, int kk) {
        async_cp16(agp + kk, &Ab[buf][ar * BK + ac]);
        async_cp16(wgp + kk,     &Bb[buf][br * BK + bc]);
        async_cp16(wgp + kk + 8, &Bb[buf][br * BK + bc + 8]);
    };

    v8f acc[4] = {v8f{}, v8f{}, v8f{}, v8f{}};

    stage(0, 0);
    wait_async();
    __syncthreads();

    int buf = 0;
    for (int kk = 0; kk < K; kk += BK) {
        if (kk + BK < K) stage(buf ^ 1, kk + BK);

        v16bf a = *(const v16bf*)&Ab[buf][(wm * 16 + (lane & 15)) * BK + klane];
        #pragma unroll
        for (int j = 0; j < 4; ++j) {
            v16bf b = *(const v16bf*)&Bb[buf][(wn * 64 + j * 16 + (lane & 15)) * BK + klane];
            acc[j] = wmma_bf16(a, b, acc[j]);
        }
        wait_async();
        __syncthreads();
        buf ^= 1;
    }

    #pragma unroll
    for (int j = 0; j < 4; ++j) {
        #pragma unroll
        for (int r = 0; r < 8; ++r) {
            int Mi = m0 + wm * 16 + r + ((lane >> 4) << 3);
            int Ni = n0 + wn * 64 + j * 16 + (lane & 15);
            float v = acc[j][r] + bias[Ni];
            if (transpose_out) {
                int bb = Mi / seq, ss = Mi % seq;
                out[((size_t)bb * N + Ni) * seq + ss] = (__bf16)v;
            } else {
                out[(size_t)Mi * N + Ni] = (__bf16)v;
            }
        }
    }
}

// ---------------------------------------------------------------------------
// Scores: Sb[b,q,k] = scale * sum_f Q[b,q,f]*K[b,k,f], masked (causal + pad).
// 8 waves per block (q strip of 128); each wave: 16 q x 64 k (4 accums),
// Q fragment reused 4x; above-diagonal 16-wide sub-tiles skipped (uniform).
// ---------------------------------------------------------------------------
__global__ __launch_bounds__(256) void sa_wmma_scores(
        const __bf16* __restrict__ Q, const __bf16* __restrict__ Km,
        const int* __restrict__ pad, float* __restrict__ Sb) {
    const int tid  = threadIdx.x;
    const int lane = tid & 31;
    const int wave = tid >> 5;
    const int k0 = blockIdx.x * 64;
    const int q0 = blockIdx.y * 128 + wave * 16;
    const int b  = blockIdx.z;
    float* sbase = Sb + (size_t)b * S_ * S_;
    const int klane = (lane >> 4) << 4;

    // #k sub-tiles (16 wide) intersecting the causal region (wave-uniform)
    int nj = (q0 + 16 - k0) >> 4;
    nj = nj < 0 ? 0 : (nj > 4 ? 4 : nj);

    v8f acc[4] = {v8f{}, v8f{}, v8f{}, v8f{}};
    if (nj > 0) {
        const __bf16* qp = Q  + ((size_t)b * S_ + q0 + (lane & 15)) * F_ + klane;
        const __bf16* kp = Km + ((size_t)b * S_ + k0 + (lane & 15)) * F_ + klane;
        for (int f = 0; f < F_; f += 32) {
            v16bf a = *(const v16bf*)qp;
            #pragma unroll
            for (int j = 0; j < 4; ++j) {
                if (j < nj) {
                    v16bf bb = *(const v16bf*)(kp + (size_t)j * 16 * F_);
                    acc[j] = wmma_bf16(a, bb, acc[j]);
                }
            }
            qp += 32; kp += 32;
        }
    }

    const float scale = 0.03125f;  // 1/sqrt(1024)
    #pragma unroll
    for (int j = 0; j < 4; ++j) {
        #pragma unroll
        for (int r = 0; r < 8; ++r) {
            int q = q0 + r + ((lane >> 4) << 3);
            int k = k0 + j * 16 + (lane & 15);
            float s = (j < nj) ? acc[j][r] * scale : NEGV;
            if (k > q || pad[b * S_ + k] == 0) s = NEGV;
            sbase[(size_t)q * S_ + k] = s;
        }
    }
}

// ---------------------------------------------------------------------------
// Per-row softmax stats: m = max, invl = 1/sum(exp(s - m)). One block per row.
// ---------------------------------------------------------------------------
__global__ void sa_row_stats(const float* __restrict__ Sb,
                             float* __restrict__ mrow, float* __restrict__ invl) {
    __shared__ float red[256];
    const int row = blockIdx.x;          // row = b*S + q
    const int tid = threadIdx.x;
    const float* sp = Sb + (size_t)row * S_;

    float m = -3.4e38f;
    for (int k = tid; k < S_; k += 256) m = fmaxf(m, sp[k]);
    red[tid] = m; __syncthreads();
    for (int o = 128; o > 0; o >>= 1) {
        if (tid < o) red[tid] = fmaxf(red[tid], red[tid + o]);
        __syncthreads();
    }
    m = red[0]; __syncthreads();

    float l = 0.f;
    for (int k = tid; k < S_; k += 256) l += __expf(sp[k] - m);
    red[tid] = l; __syncthreads();
    for (int o = 128; o > 0; o >>= 1) {
        if (tid < o) red[tid] += red[tid + o];
        __syncthreads();
    }
    if (tid == 0) { mrow[row] = m; invl[row] = 1.0f / red[0]; }
}

// ---------------------------------------------------------------------------
// PV: out[b,q,f] = sum_k softmax(S)[q,k] * V[k,f]; prob built on the fly
// (exp -> bf16) into the A fragment, reused across 4 f-tiles (exp cost / 4).
// Vt is [b, f, s] so the B operand is contiguous. Causal loop bound.
// ---------------------------------------------------------------------------
__global__ __launch_bounds__(256) void sa_wmma_pv(
        const float* __restrict__ Sb, const __bf16* __restrict__ Vt,
        const float* __restrict__ mrow, const float* __restrict__ invl,
        float* __restrict__ out) {
    const int tid  = threadIdx.x;
    const int lane = tid & 31;
    const int wave = tid >> 5;
    const int f0 = blockIdx.x * 64;
    const int q0 = blockIdx.y * 128 + wave * 16;
    const int b  = blockIdx.z;
    const int q  = q0 + (lane & 15);
    const int klane = (lane >> 4) << 4;

    const float m  = mrow[b * S_ + q];
    const float il = invl[b * S_ + q];
    const float*  sp = Sb + ((size_t)b * S_ + q) * S_ + klane;
    const __bf16* vp = Vt + ((size_t)b * F_ + f0 + (lane & 15)) * S_ + klane;

    v8f acc[4] = {v8f{}, v8f{}, v8f{}, v8f{}};
    const int kend = q0 + 16;   // last key tile intersecting the causal region
    for (int k0 = 0; k0 < kend; k0 += 32) {
        v16bf a;
        #pragma unroll
        for (int i = 0; i < 16; ++i)
            a[i] = (__bf16)(__expf(sp[i] - m) * il);
        #pragma unroll
        for (int j = 0; j < 4; ++j) {
            v16bf bb = *(const v16bf*)(vp + (size_t)j * 16 * S_);
            acc[j] = wmma_bf16(a, bb, acc[j]);
        }
        sp += 32; vp += 32;
    }

    #pragma unroll
    for (int j = 0; j < 4; ++j) {
        #pragma unroll
        for (int r = 0; r < 8; ++r) {
            int qq = q0 + r + ((lane >> 4) << 3);
            out[((size_t)b * S_ + qq) * F_ + f0 + j * 16 + (lane & 15)] = acc[j][r];
        }
    }
}

// ---------------------------------------------------------------------------
extern "C" void kernel_launch(void* const* d_in, const int* in_sizes, int n_in,
                              void* d_out, int out_size, void* d_ws, size_t ws_size,
                              hipStream_t stream) {
    (void)in_sizes; (void)n_in; (void)out_size; (void)ws_size;

    const float* x   = (const float*)d_in[0];
    // d_in[1] = attn_mask: causal, computed analytically (k > q) — not read.
    const int*   pad = (const int*)d_in[2];
    const float* Wq  = (const float*)d_in[3];
    const float* bq  = (const float*)d_in[4];
    const float* Wk  = (const float*)d_in[5];
    const float* bk  = (const float*)d_in[6];
    const float* Wv  = (const float*)d_in[7];
    const float* bv  = (const float*)d_in[8];
    float* out = (float*)d_out;

    // Workspace carve-up (~267 MB): scores (128 MB) live in the 192 MB L2.
    char* ws = (char*)d_ws;
    size_t off = 0;
    auto carve = [&](size_t bytes) -> void* {
        void* p = ws + off;
        off += (bytes + 255) & ~(size_t)255;
        return p;
    };
    const size_t NT = (size_t)B_ * S_ * F_;     // 16.8M elems
    __bf16* xb   = (__bf16*)carve(NT * 2);
    __bf16* wqb  = (__bf16*)carve((size_t)F_ * F_ * 2);
    __bf16* wkb  = (__bf16*)carve((size_t)F_ * F_ * 2);
    __bf16* wvb  = (__bf16*)carve((size_t)F_ * F_ * 2);
    __bf16* Qb   = (__bf16*)carve(NT * 2);
    __bf16* Kb   = (__bf16*)carve(NT * 2);
    __bf16* Vtb  = (__bf16*)carve(NT * 2);      // [b, f, s]
    float*  Sbuf = (float*) carve((size_t)B_ * S_ * S_ * 4);
    float*  mbuf = (float*) carve((size_t)B_ * S_ * 4);
    float*  lbuf = (float*) carve((size_t)B_ * S_ * 4);

    // 1) fp32 -> bf16
    sa_cvt_bf16<<<4096, 256, 0, stream>>>(x,  xb,  (int)NT);
    sa_cvt_bf16<<<1024, 256, 0, stream>>>(Wq, wqb, F_ * F_);
    sa_cvt_bf16<<<1024, 256, 0, stream>>>(Wk, wkb, F_ * F_);
    sa_cvt_bf16<<<1024, 256, 0, stream>>>(Wv, wvb, F_ * F_);

    // 2) QKV projections (async-LDS staged WMMA); V stored transposed.
    dim3 gproj(F_ / BN, (B_ * S_) / BM);        // (8, 256)
    sa_wmma_linear<<<gproj, 256, 0, stream>>>(xb, wqb, bq, Qb,  F_, F_, 0, S_);
    sa_wmma_linear<<<gproj, 256, 0, stream>>>(xb, wkb, bk, Kb,  F_, F_, 0, S_);
    sa_wmma_linear<<<gproj, 256, 0, stream>>>(xb, wvb, bv, Vtb, F_, F_, 1, S_);

    // 3) masked scaled scores (WMMA, causal sub-tile skip), fp32 to L2
    sa_wmma_scores<<<dim3(S_ / 64, S_ / 128, B_), 256, 0, stream>>>(Qb, Kb, pad, Sbuf);

    // 4) row max / inverse sum-exp (LDS reduction)
    sa_row_stats<<<B_ * S_, 256, 0, stream>>>(Sbuf, mbuf, lbuf);

    // 5) prob @ V (WMMA), exp fused into A-fragment build
    sa_wmma_pv<<<dim3(F_ / 64, S_ / 128, B_), 256, 0, stream>>>(Sbuf, Vtb, mbuf, lbuf, out);
}
